// CoreAttention_21947282882676
// MI455X (gfx1250) — matrix operand order repaired and checked by reading, not compiled, not run on hardware
//
#include <hip/hip_runtime.h>

typedef __bf16 bf16_t;
typedef __attribute__((ext_vector_type(16))) __bf16 v16bf;
typedef __attribute__((ext_vector_type(8)))  __bf16 v8bf;
typedef __attribute__((ext_vector_type(2)))  __bf16 v2bf;
typedef __attribute__((ext_vector_type(8)))  float  v8f;

#define SQ     2048
#define DHEAD  128
#define NHEAD  32
#define NBATCH 2
#define QBLK   128   // query rows per workgroup
#define KB     32    // keys per iteration
#define NWAVES 8

#define NEG_INF (-__builtin_inff())
// scores * (1/(sqrt(128)*2)) * 2 == scores / sqrt(128); fold in log2(e) and use exp2
#define SCALE2 (0.08838834764831845f * 1.4426950408889634f)

#if defined(__has_builtin)
#if __has_builtin(__builtin_amdgcn_permlane16)
#define USE_PERMLANE 1
#endif
#if __has_builtin(__builtin_amdgcn_exp2f)
#define EXP2F(x) __builtin_amdgcn_exp2f(x)
#endif
#endif
#ifndef EXP2F
#define EXP2F(x) exp2f(x)
#endif

#ifdef USE_PERMLANE
__device__ __forceinline__ float permx(float x, unsigned lo, unsigned hi) {
    unsigned u = __float_as_uint(x);
    return __uint_as_float(__builtin_amdgcn_permlane16(u, u, lo, hi, false, false));
}
#endif

// max-butterfly across the 16 lanes of each half-wave (pure VALU via v_permlane16)
__device__ __forceinline__ float xmax16(float v) {
#ifdef USE_PERMLANE
    v = fmaxf(v, permx(v, 0x67452301u, 0xEFCDAB89u));   // xor 1
    v = fmaxf(v, permx(v, 0x54761032u, 0xDCFE98BAu));   // xor 2
    v = fmaxf(v, permx(v, 0x32107654u, 0xBA98FEDCu));   // xor 4
    v = fmaxf(v, permx(v, 0xFEDCBA98u, 0x76543210u));   // xor 8
#else
    v = fmaxf(v, __shfl_xor(v, 1, 16));
    v = fmaxf(v, __shfl_xor(v, 2, 16));
    v = fmaxf(v, __shfl_xor(v, 4, 16));
    v = fmaxf(v, __shfl_xor(v, 8, 16));
#endif
    return v;
}

__device__ __forceinline__ v8bf cvt8(float4 a, float4 b) {
    v8bf r;
    r[0]=(bf16_t)a.x; r[1]=(bf16_t)a.y; r[2]=(bf16_t)a.z; r[3]=(bf16_t)a.w;
    r[4]=(bf16_t)b.x; r[5]=(bf16_t)b.y; r[6]=(bf16_t)b.z; r[7]=(bf16_t)b.w;
    return r;
}
__device__ __forceinline__ v16bf cvt16(float4 a0, float4 a1, float4 a2, float4 a3) {
    v16bf r;
    r[0]=(bf16_t)a0.x;  r[1]=(bf16_t)a0.y;  r[2]=(bf16_t)a0.z;  r[3]=(bf16_t)a0.w;
    r[4]=(bf16_t)a1.x;  r[5]=(bf16_t)a1.y;  r[6]=(bf16_t)a1.z;  r[7]=(bf16_t)a1.w;
    r[8]=(bf16_t)a2.x;  r[9]=(bf16_t)a2.y;  r[10]=(bf16_t)a2.z; r[11]=(bf16_t)a2.w;
    r[12]=(bf16_t)a3.x; r[13]=(bf16_t)a3.y; r[14]=(bf16_t)a3.z; r[15]=(bf16_t)a3.w;
    return r;
}

struct TrueT  { static constexpr bool value = true;  };
struct FalseT { static constexpr bool value = false; };

__launch_bounds__(256, 1)
__global__ void fa_fwd_causal(const float* __restrict__ Q,
                              const float* __restrict__ K,
                              const float* __restrict__ V,
                              float* __restrict__ Out)
{
    // Double-buffered LDS tiles: K as bf16 [key][d], V as bf16 transposed [d][key].
    __shared__ __align__(32) bf16_t Ks[2][KB * DHEAD];       // 16 KB
    __shared__ __align__(32) bf16_t VT[2][DHEAD * KB];       // 16 KB
    __shared__ __align__(32) bf16_t Ps[NWAVES][16 * KB];     //  8 KB

    const int tid  = threadIdx.x;
    const int w    = __builtin_amdgcn_readfirstlane(tid >> 5);  // scalar wave id
    const int lane = tid & 31;
    const int ln16 = lane & 15;
    const int hi   = lane >> 4;      // half-wave select

    const int qblk = blockIdx.x;
    const int bh   = blockIdx.y;     // b*NHEAD + h
    const int b    = bh / NHEAD;
    const int h    = bh % NHEAD;

    const size_t base = (size_t)bh * SQ * DHEAD;
    const float* Qb = Q + base;
    const float* Kb = K + base;
    const float* Vb = V + base;

    const int q0 = qblk * QBLK + w * 16;   // scalar: first query row of this wave

    // ---- preload Q tile (16 x 128) into WMMA A-layout registers, fp32 -> bf16 ----
    const int aqrow = q0 + ln16;
    v16bf Qa[4];
#pragma unroll
    for (int c = 0; c < 4; ++c) {
        const float* s1 = Qb + (size_t)aqrow * DHEAD + c * 32 + hi * 16;
        const float* s2 = s1 + 8;
        Qa[c] = cvt16(((const float4*)s1)[0], ((const float4*)s1)[1],
                      ((const float4*)s2)[0], ((const float4*)s2)[1]);
    }

    // B-operand of all-ones (32x16) for WMMA-based row sums
    v16bf onesB;
#pragma unroll
    for (int i = 0; i < 16; ++i) onesB[i] = (bf16_t)1.0f;

    // ---- accumulators ----
    v8f O[8];
#pragma unroll
    for (int dt = 0; dt < 8; ++dt) { v8f z = {}; O[dt] = z; }
    float m_run[8], l_run[8];
#pragma unroll
    for (int r = 0; r < 8; ++r) { m_run[r] = NEG_INF; l_run[r] = 0.0f; }

    // ---- staging ownership: 2 adjacent keys x 8 d-values per thread ----
    const int skey = (tid & 15) * 2;     // 0,2,...,30
    const int sd   = (tid >> 4) * 8;     // 0,8,...,120
    float4 k0a, k0b, k1a, k1b, v0a, v0b, v1a, v1b;

    auto gload = [&](int kb) {
        const float* kp0 = Kb + (size_t)(kb + skey) * DHEAD + sd;
        const float* kp1 = kp0 + DHEAD;
        const float* vp0 = Vb + (size_t)(kb + skey) * DHEAD + sd;
        const float* vp1 = vp0 + DHEAD;
        k0a = ((const float4*)kp0)[0]; k0b = ((const float4*)kp0)[1];
        k1a = ((const float4*)kp1)[0]; k1b = ((const float4*)kp1)[1];
        v0a = ((const float4*)vp0)[0]; v0b = ((const float4*)vp0)[1];
        v1a = ((const float4*)vp1)[0]; v1b = ((const float4*)vp1)[1];
    };
    auto lstore = [&](int buf) {
        *(v8bf*)&Ks[buf][(skey)     * DHEAD + sd] = cvt8(k0a, k0b);
        *(v8bf*)&Ks[buf][(skey + 1) * DHEAD + sd] = cvt8(k1a, k1b);
        const float va[8] = { v0a.x, v0a.y, v0a.z, v0a.w, v0b.x, v0b.y, v0b.z, v0b.w };
        const float vb[8] = { v1a.x, v1a.y, v1a.z, v1a.w, v1b.x, v1b.y, v1b.z, v1b.w };
#pragma unroll
        for (int j = 0; j < 8; ++j) {
            v2bf t; t[0] = (bf16_t)va[j]; t[1] = (bf16_t)vb[j];
            *(v2bf*)&VT[buf][(sd + j) * KB + skey] = t;   // packed 2-key dword store
        }
    };

    const int nkb = qblk * 4 + 4;   // causal: key blocks with kb*32 <= qblk*128+127

    // prologue: stage tile 0
    gload(0);
    lstore(0);
    __syncthreads();

    for (int it = 0; it < nkb; ++it) {
        const int kb = it * KB;

        // issue next tile's global loads now; latency hides under compute
        if (it + 1 < nkb) gload((it + 1) * KB);

        if (kb <= q0) {   // scalar causal skip
            const bf16_t* ksb = Ks[it & 1];
            const bf16_t* vtb = VT[it & 1];

            auto body = [&](auto D) {
                constexpr bool diag = decltype(D)::value;

                // ---- S = Q * K^T : two 16x16 tiles over 32 keys, K-dim = 128 ----
                v8f S0 = {}, S1 = {};
#pragma unroll
                for (int c = 0; c < 4; ++c) {
                    v16bf bk0 = *(const v16bf*)&ksb[(ln16)      * DHEAD + c * 32 + hi * 16];
                    v16bf bk1 = *(const v16bf*)&ksb[(16 + ln16) * DHEAD + c * 32 + hi * 16];
                    S0 = __builtin_amdgcn_wmma_f32_16x16x32_bf16(false, Qa[c], false, bk0, (short)0, S0, false, false);
                    S1 = __builtin_amdgcn_wmma_f32_16x16x32_bf16(false, Qa[c], false, bk1, (short)0, S1, false, false);
                }

                // ---- scale (log2 domain), diagonal-only mask, row max + exp2 ----
                const int kcol0 = kb + ln16;
                const int kcol1 = kb + 16 + ln16;
                float fscale[8];
#pragma unroll
                for (int r = 0; r < 8; ++r) {
                    const int q = q0 + r + 8 * hi;
                    float s0 = S0[r] * SCALE2;
                    float s1 = S1[r] * SCALE2;
                    if (diag) {
                        if (kcol0 > q) s0 = NEG_INF;
                        if (kcol1 > q) s1 = NEG_INF;
                    }
                    const float mr = xmax16(fmaxf(s0, s1));
                    const float nm = fmaxf(m_run[r], mr);
                    S0[r] = EXP2F(s0 - nm);
                    S1[r] = EXP2F(s1 - nm);
                    fscale[r] = EXP2F(m_run[r] - nm);
                    m_run[r]  = nm;
                }
#pragma unroll
                for (int dt = 0; dt < 8; ++dt)
#pragma unroll
                    for (int r = 0; r < 8; ++r) O[dt][r] *= fscale[r];

                // ---- P (C layout, fp32) -> bf16 A layout via per-wave LDS bounce ----
#pragma unroll
                for (int r = 0; r < 8; ++r) {
                    Ps[w][(r + 8 * hi) * KB + ln16]      = (bf16_t)S0[r];
                    Ps[w][(r + 8 * hi) * KB + 16 + ln16] = (bf16_t)S1[r];
                }
                // same-wave DS ops are in-order: no barrier needed
                v8bf plo = *(const v8bf*)&Ps[w][ln16 * KB + hi * 16];
                v8bf phi = *(const v8bf*)&Ps[w][ln16 * KB + hi * 16 + 8];
                v16bf Pa;
#pragma unroll
                for (int i = 0; i < 8; ++i) { Pa[i] = plo[i]; Pa[i + 8] = phi[i]; }

                // ---- row sums via WMMA: Pa x ones -> every column holds the row sum ----
                v8f Ssum = {};
                Ssum = __builtin_amdgcn_wmma_f32_16x16x32_bf16(false, Pa, false, onesB, (short)0, Ssum, false, false);
#pragma unroll
                for (int r = 0; r < 8; ++r) l_run[r] = l_run[r] * fscale[r] + Ssum[r];

                // ---- O += P * V : 8 d-tiles, K-dim = 32 keys ----
#pragma unroll
                for (int dt = 0; dt < 8; ++dt) {
                    v16bf vbm = *(const v16bf*)&vtb[(dt * 16 + ln16) * KB + hi * 16];
                    O[dt] = __builtin_amdgcn_wmma_f32_16x16x32_bf16(false, Pa, false, vbm, (short)0, O[dt], false, false);
                }
            };

            if (kb == q0) body(TrueT{}); else body(FalseT{});
        }

        // convert + store next tile into the other LDS buffer, then one barrier
        if (it + 1 < nkb) lstore((it + 1) & 1);
        __syncthreads();
    }

    // ---- epilogue: divide by softmax denominator, write [b, sq, h*d] ----
    float inv[8];
#pragma unroll
    for (int r = 0; r < 8; ++r) inv[r] = 1.0f / l_run[r];
#pragma unroll
    for (int dt = 0; dt < 8; ++dt) {
#pragma unroll
        for (int r = 0; r < 8; ++r) {
            const int q    = q0 + r + 8 * hi;
            const int dcol = dt * 16 + ln16;
            Out[(size_t)(b * SQ + q) * (NHEAD * DHEAD) + h * DHEAD + dcol] = O[dt][r] * inv[r];
        }
    }
}

extern "C" void kernel_launch(void* const* d_in, const int* in_sizes, int n_in,
                              void* d_out, int out_size, void* d_ws, size_t ws_size,
                              hipStream_t stream) {
    (void)in_sizes; (void)n_in; (void)out_size; (void)d_ws; (void)ws_size;
    const float* Q = (const float*)d_in[0];
    const float* K = (const float*)d_in[1];
    const float* V = (const float*)d_in[2];
    float* O = (float*)d_out;
    dim3 grid(SQ / QBLK, NBATCH * NHEAD);
    hipLaunchKernelGGL(fa_fwd_causal, grid, dim3(256), 0, stream, Q, K, V, O);
}
